// LstmCRF_29051158790403
// MI455X (gfx1250) — matrix operand order repaired and checked by reading, not compiled, not run on hardware
//
#include <hip/hip_runtime.h>
#include <hip/hip_bf16.h>

// ---------------------------------------------------------------------------
// Problem dimensions (fixed by the reference)
// ---------------------------------------------------------------------------
constexpr int Bc  = 256;     // batch
constexpr int Tc  = 128;     // time steps
constexpr int Dc  = 256;     // embedding dim
constexpr int Hc  = 256;     // hidden per direction
constexpr int FHc = 4 * Hc;  // 1024 gate columns
constexpr int Kc  = 9;       // tag count

typedef __attribute__((ext_vector_type(16))) __bf16 v16bf;
typedef __attribute__((ext_vector_type(8)))  float  v8f;

union FragBF { v16bf v; uint4 q[2]; };
union FragF  { v8f   v; float4 f4[2]; };

// A-fragment (16x32 bf16, M x K): lane<16 row M=idx holds K {0..7}U{16..23},
// lane>=16 holds K {8..15}U{24..31}. rowptr already offset to k-base.
__device__ __forceinline__ v16bf load_a_frag(const __bf16* rowptr, unsigned half) {
    FragBF f;
    const uint4* p = reinterpret_cast<const uint4*>(rowptr);
    f.q[0] = p[half];
    f.q[1] = p[2 + half];
    return f.v;
}

// B-fragment (32x16 bf16, K x N), B[k][n] = W[n][k] from row-major W:
// lane half 0 holds K {0..7}U{16..23}, half 1 holds K {8..15}U{24..31}.
__device__ __forceinline__ v16bf load_b_frag(const __bf16* rowptr, unsigned half) {
    FragBF f;
    const uint4* p = reinterpret_cast<const uint4*>(rowptr);
    f.q[0] = p[half];
    f.q[1] = p[2 + half];
    return f.v;
}

__device__ __forceinline__ float sigf(float x) { return 1.0f / (1.0f + __expf(-x)); }

// ---------------------------------------------------------------------------
// fp32 -> bf16 weight cast
// ---------------------------------------------------------------------------
__global__ __launch_bounds__(256) void cast_bf16_k(const float* __restrict__ src,
                                                   __bf16* __restrict__ dst, int n) {
    int gid = blockIdx.x * 256 + threadIdx.x;
    if (gid < n) dst[gid] = (__bf16)src[gid];
}

// ---------------------------------------------------------------------------
// Embedding gather -> bf16 x [B][T][D]
// ---------------------------------------------------------------------------
__global__ __launch_bounds__(256) void embed_k(const int* __restrict__ ix,
                                               const float* __restrict__ emb,
                                               __bf16* __restrict__ xout) {
    int gid = blockIdx.x * 256 + threadIdx.x;   // exactly B*T*D threads
    int bt  = gid >> 8;                         // D == 256
    int d   = gid & 255;
    int tok = ix[bt];
    xout[gid] = (__bf16)emb[(size_t)tok * Dc + d];
}

// ---------------------------------------------------------------------------
// Input projection GEMM:
//   Xp_frag[t][tb][nt][lane][r] = bias + x[b][t][:] . Wih[n][:]
// M dimension ordered (t, b): M-tile = 16 batch rows at fixed t, so the
// D-fragment layout written here is exactly what the recurrent kernel reads.
// Throughput kernel; 4096 blocks hide the load latency via occupancy.
// ---------------------------------------------------------------------------
__global__ __launch_bounds__(256) void inproj_gemm_k(
    const __bf16* __restrict__ X,        // [B][T][D]
    const __bf16* __restrict__ Wf, const float* __restrict__ bf,
    const __bf16* __restrict__ Wb, const float* __restrict__ bb,
    float* __restrict__ Xpf, float* __restrict__ Xpb) {

    const __bf16* W    = (blockIdx.y == 0) ? Wf  : Wb;
    const float*  bias = (blockIdx.y == 0) ? bf  : bb;
    float*        Xp   = (blockIdx.y == 0) ? Xpf : Xpb;

    const unsigned tid  = threadIdx.x;
    const unsigned lane = tid & 31u, wave = tid >> 5;
    const unsigned half = lane >> 4, idx = lane & 15u;
    const unsigned mbase = blockIdx.x * 16;          // m = (t<<8) | b
    const unsigned t  = mbase >> 8;
    const unsigned tb = (mbase >> 4) & 15u;

    v8f acc[8];
#pragma unroll
    for (int ntl = 0; ntl < 8; ++ntl) {
        unsigned n = (wave * 8 + ntl) * 16 + idx;
        float bv = bias[n];
#pragma unroll
        for (int r = 0; r < 8; ++r) acc[ntl][r] = bv;
    }

    // A row: batch b = (mbase & 255) + idx at time t
    const __bf16* arow = X + ((size_t)((mbase & 255u) + idx) * Tc + t) * Dc;

#pragma unroll
    for (int kk = 0; kk < 8; ++kk) {
        v16bf a = load_a_frag(arow + kk * 32, half);
        v16bf b[8];
#pragma unroll
        for (int ntl = 0; ntl < 8; ++ntl) {
            unsigned n = (wave * 8 + ntl) * 16 + idx;
            b[ntl] = load_b_frag(W + (size_t)n * Dc + kk * 32, half);
        }
#pragma unroll
        for (int ntl = 0; ntl < 8; ++ntl) {
            acc[ntl] = __builtin_amdgcn_wmma_f32_16x16x32_bf16(
                false, a, false, b[ntl], (short)0, acc[ntl], false, false);
        }
    }

    // fragment-packed store: [t][tb][nt][lane][r], two b128 per fragment
#pragma unroll
    for (int ntl = 0; ntl < 8; ++ntl) {
        unsigned nt = wave * 8 + ntl;
        float* outp = Xp + ((((size_t)t * 16 + tb) * 64 + nt) * 32 + lane) * 8;
        FragF fo; fo.v = acc[ntl];
        *reinterpret_cast<float4*>(outp)     = fo.f4[0];
        *reinterpret_cast<float4*>(outp + 4) = fo.f4[1];
    }
}

// ---------------------------------------------------------------------------
// Recurrent LSTM.  One block owns a 16-row batch tile for all T steps.
// Whh is loop-invariant:
//   * k =   0..127 : 32 B-fragments persist in VGPRs (loaded once)
//   * k = 128..255 : 256KB LDS-resident (loaded once, ds_load_b128 per step)
// => zero Whh global traffic inside the sequential t-loop.
// Wave w owns hidden tiles {2w,2w+1} of every gate; c stays in registers,
// h (16x256 bf16) in LDS.  Dynamic LDS: 8KB h + 256KB Whh-hi = 264KB.
// blockIdx.y selects direction (0=forward, 1=backward).
// ---------------------------------------------------------------------------
__global__ __launch_bounds__(256, 1) void lstm_k(
    const float* __restrict__ Xpf, const __bf16* __restrict__ Whhf,
    const float* __restrict__ h0f, const float* __restrict__ c0f,
    float* __restrict__ houtf,
    const float* __restrict__ Xpb, const __bf16* __restrict__ Whhb,
    const float* __restrict__ h0b, const float* __restrict__ c0b,
    float* __restrict__ houtb) {

    const bool fwd = (blockIdx.y == 0);
    const float*  Xp   = fwd ? Xpf  : Xpb;
    const __bf16* Whh  = fwd ? Whhf : Whhb;
    const float*  h0   = fwd ? h0f  : h0b;
    const float*  c0   = fwd ? c0f  : c0b;
    float*        hout = fwd ? houtf : houtb;
    const int t0 = fwd ? 0 : (Tc - 1);
    const int dt = fwd ? 1 : -1;

    extern __shared__ __align__(16) char smem[];
    __bf16 (*h_sh)[Hc] = reinterpret_cast<__bf16(*)[Hc]>(smem);          // 8 KB
    __bf16* w_hi = reinterpret_cast<__bf16*>(smem + 16 * Hc * 2);        // [1024][128]

    const unsigned tid  = threadIdx.x;
    const unsigned lane = tid & 31u, wave = tid >> 5;
    const unsigned half = lane >> 4, idx = lane & 15u;
    const unsigned tb   = blockIdx.x;             // batch tile
    const unsigned bbase = tb * 16;

    // ---- one-time staging -------------------------------------------------
    for (unsigned i = tid; i < 16u * Hc; i += 256u) {
        unsigned row = i >> 8, j = i & 255u;
        h_sh[row][j] = (__bf16)h0[(size_t)(bbase + row) * Hc + j];
    }
    // Whh k-high half (k=128..255) -> LDS, 16384 uint4
    {
        const uint4* wsrc = reinterpret_cast<const uint4*>(Whh); // 32 uint4 per row
        uint4*       wdst = reinterpret_cast<uint4*>(w_hi);      // 16 uint4 per row
        for (unsigned u = tid; u < 16384u; u += 256u) {
            unsigned n = u >> 4, seg = u & 15u;
            wdst[u] = wsrc[n * 32 + 16 + seg];
        }
    }
    // Whh k-low half (k=0..127) -> persistent register fragments
    v16bf blo[32];                                // [kk 0..3][tile q 0..7]
#pragma unroll
    for (int kk = 0; kk < 4; ++kk) {
#pragma unroll
        for (int q = 0; q < 8; ++q) {
            unsigned n = (unsigned)(((q >> 1) * 16 + wave * 2 + (q & 1)) * 16) + idx;
            blo[kk * 8 + q] = load_b_frag(Whh + (size_t)n * Hc + kk * 32, half);
        }
    }
    float creg[2][8];
#pragma unroll
    for (int p = 0; p < 2; ++p) {
        unsigned j = (wave * 2 + p) * 16 + idx;
#pragma unroll
        for (int r = 0; r < 8; ++r)
            creg[p][r] = c0[(size_t)(bbase + r + half * 8) * Hc + j];
    }
    __syncthreads();

    // ---- sequential scan --------------------------------------------------
    for (int s = 0; s < Tc; ++s) {
        const int t = t0 + s * dt;

        // prefetch next timestep's 64KB Xp slice for this batch tile
        if (s + 1 < Tc) {
            const float* pf =
                Xp + (((size_t)(t + dt) * 16 + tb) * 64) * 256 + (size_t)tid * 64;
            __builtin_prefetch(pf, 0, 3);
        }

        // accumulator init: fragment-packed Xp, two b128 per fragment
        v8f acc[4][2];
#pragma unroll
        for (int g = 0; g < 4; ++g) {
#pragma unroll
            for (int p = 0; p < 2; ++p) {
                unsigned nt = (unsigned)(g * 16 + wave * 2 + p);
                const float* ip =
                    Xp + ((((size_t)t * 16 + tb) * 64 + nt) * 32 + lane) * 8;
                FragF fi;
                fi.f4[0] = *reinterpret_cast<const float4*>(ip);
                fi.f4[1] = *reinterpret_cast<const float4*>(ip + 4);
                acc[g][p] = fi.v;
            }
        }

        // k = 0..127 : B operands already in registers
#pragma unroll
        for (int kk = 0; kk < 4; ++kk) {
            v16bf a = load_a_frag(&h_sh[idx][kk * 32], half);
#pragma unroll
            for (int q = 0; q < 8; ++q) {
                int g = q >> 1, p = q & 1;
                acc[g][p] = __builtin_amdgcn_wmma_f32_16x16x32_bf16(
                    false, a, false, blo[kk * 8 + q], (short)0, acc[g][p],
                    false, false);
            }
        }
        // k = 128..255 : B operands from LDS-resident Whh-hi
#pragma unroll
        for (int kk = 0; kk < 4; ++kk) {
            v16bf a = load_a_frag(&h_sh[idx][128 + kk * 32], half);
#pragma unroll
            for (int q = 0; q < 8; ++q) {
                int g = q >> 1, p = q & 1;
                unsigned n =
                    (unsigned)(((q >> 1) * 16 + wave * 2 + (q & 1)) * 16) + idx;
                v16bf b = load_b_frag(w_hi + (size_t)n * 128 + kk * 32, half);
                acc[g][p] = __builtin_amdgcn_wmma_f32_16x16x32_bf16(
                    false, a, false, b, (short)0, acc[g][p], false, false);
            }
        }
        __syncthreads();   // all h_sh reads complete before overwrite

#pragma unroll
        for (int p = 0; p < 2; ++p) {
            unsigned j = (wave * 2 + p) * 16 + idx;
#pragma unroll
            for (int r = 0; r < 8; ++r) {
                float gi = acc[0][p][r], gf = acc[1][p][r];
                float gg = acc[2][p][r], go = acc[3][p][r];
                float c = sigf(gf) * creg[p][r] + sigf(gi) * tanhf(gg);
                creg[p][r] = c;
                float h = sigf(go) * tanhf(c);
                unsigned row = r + half * 8;
                h_sh[row][j] = (__bf16)h;
                hout[((size_t)(bbase + row) * Tc + t) * Hc + j] = h;
            }
        }
        __syncthreads();
    }
}

// ---------------------------------------------------------------------------
// Emission: emis[b,t,k] = b_emit[k] + hf.W[k,:H] + hb.W[k,H:]
// ---------------------------------------------------------------------------
__global__ __launch_bounds__(256) void emis_k(
    const float* __restrict__ hf, const float* __restrict__ hb,
    const float* __restrict__ We, const float* __restrict__ be,
    float* __restrict__ emis) {
    int gid = blockIdx.x * 256 + threadIdx.x;
    if (gid >= Bc * Tc * Kc) return;
    int k = gid % Kc, bt = gid / Kc;
    const float* wf  = We + (size_t)k * (2 * Hc);
    const float* hfr = hf + (size_t)bt * Hc;
    const float* hbr = hb + (size_t)bt * Hc;
    float sum = be[k];
    for (int j = 0; j < Hc; ++j) sum += hfr[j] * wf[j] + hbr[j] * wf[Hc + j];
    emis[gid] = sum;
}

// ---------------------------------------------------------------------------
// CRF negative log-likelihood: one thread per batch element -> partial[b]
// ---------------------------------------------------------------------------
__global__ __launch_bounds__(256) void crf_loss_k(
    const float* __restrict__ emis, const int* __restrict__ label,
    const int* __restrict__ input_x, const float* __restrict__ start_t,
    const float* __restrict__ end_t, const float* __restrict__ trans,
    float* __restrict__ partial) {
    int b = threadIdx.x;
    const float* e  = emis + (size_t)b * Tc * Kc;
    const int*   tg = label + (size_t)b * Tc;
    const int*   ix = input_x + (size_t)b * Tc;

    float sc[Kc], ns[Kc];
    int   tag0 = tg[0];
    float num  = start_t[tag0] + e[tag0];
    for (int j = 0; j < Kc; ++j) sc[j] = start_t[j] + e[j];

    int cnt = (ix[0] != 0) ? 1 : 0;
    for (int t = 1; t < Tc; ++t) {
        bool m = (ix[t] != 0);
        if (m) {
            num += trans[tg[t - 1] * Kc + tg[t]] + e[t * Kc + tg[t]];
            cnt += 1;
            for (int j = 0; j < Kc; ++j) {
                float mx = -1e30f;
                for (int i = 0; i < Kc; ++i)
                    mx = fmaxf(mx, sc[i] + trans[i * Kc + j]);
                float ssum = 0.f;
                for (int i = 0; i < Kc; ++i)
                    ssum += __expf(sc[i] + trans[i * Kc + j] - mx);
                ns[j] = mx + __logf(ssum) + e[t * Kc + j];
            }
            for (int j = 0; j < Kc; ++j) sc[j] = ns[j];
        }
    }
    int last = cnt - 1; if (last < 0) last = 0;
    num += end_t[tg[last]];

    float mx = -1e30f;
    for (int j = 0; j < Kc; ++j) mx = fmaxf(mx, sc[j] + end_t[j]);
    float ssum = 0.f;
    for (int j = 0; j < Kc; ++j) ssum += __expf(sc[j] + end_t[j] - mx);
    partial[b] = num - (mx + __logf(ssum));
}

__global__ __launch_bounds__(256) void reduce_k(const float* __restrict__ partial,
                                                float* __restrict__ out) {
    __shared__ float sm[256];
    sm[threadIdx.x] = partial[threadIdx.x];
    __syncthreads();
    for (int s = 128; s > 0; s >>= 1) {
        if ((int)threadIdx.x < s) sm[threadIdx.x] += sm[threadIdx.x + s];
        __syncthreads();
    }
    if (threadIdx.x == 0) out[0] = -sm[0];
}

// ---------------------------------------------------------------------------
// Viterbi decode (unmasked): one thread per batch element
// ---------------------------------------------------------------------------
__global__ __launch_bounds__(256) void viterbi_k(
    const float* __restrict__ emis, const float* __restrict__ start_t,
    const float* __restrict__ end_t, const float* __restrict__ trans,
    unsigned char* __restrict__ hist, float* __restrict__ out_tags) {
    int b = threadIdx.x;
    const float*   e  = emis + (size_t)b * Tc * Kc;
    unsigned char* hb = hist + (size_t)b * Tc * Kc;

    float sc[Kc], ns[Kc];
    for (int j = 0; j < Kc; ++j) sc[j] = start_t[j] + e[j];
    for (int t = 1; t < Tc; ++t) {
        for (int j = 0; j < Kc; ++j) {
            float best = sc[0] + trans[j];
            int bi = 0;
            for (int i = 1; i < Kc; ++i) {
                float v = sc[i] + trans[i * Kc + j];
                if (v > best) { best = v; bi = i; }
            }
            ns[j] = best + e[t * Kc + j];
            hb[t * Kc + j] = (unsigned char)bi;
        }
        for (int j = 0; j < Kc; ++j) sc[j] = ns[j];
    }
    int tag = 0; float best = sc[0] + end_t[0];
    for (int j = 1; j < Kc; ++j) {
        float v = sc[j] + end_t[j];
        if (v > best) { best = v; tag = j; }
    }
    out_tags[(size_t)b * Tc + (Tc - 1)] = (float)tag;
    for (int t = Tc - 1; t >= 1; --t) {
        tag = hb[t * Kc + tag];
        out_tags[(size_t)b * Tc + (t - 1)] = (float)tag;
    }
}

// ---------------------------------------------------------------------------
// Host launcher
// ---------------------------------------------------------------------------
extern "C" void kernel_launch(void* const* d_in, const int* in_sizes, int n_in,
                              void* d_out, int out_size, void* d_ws, size_t ws_size,
                              hipStream_t stream) {
    const int*   input_x = (const int*)d_in[0];
    const int*   label   = (const int*)d_in[1];
    const float* emb     = (const float*)d_in[2];
    const float* Wih_f   = (const float*)d_in[3];
    const float* Whh_f   = (const float*)d_in[4];
    const float* b_f     = (const float*)d_in[5];
    const float* Wih_b   = (const float*)d_in[6];
    const float* Whh_b   = (const float*)d_in[7];
    const float* b_b     = (const float*)d_in[8];
    const float* W_emit  = (const float*)d_in[9];
    const float* b_emit  = (const float*)d_in[10];
    const float* start_t = (const float*)d_in[11];
    const float* end_t   = (const float*)d_in[12];
    const float* trans   = (const float*)d_in[13];
    const float* h0f     = (const float*)d_in[14];
    const float* c0f     = (const float*)d_in[15];
    const float* h0b     = (const float*)d_in[16];
    const float* c0b     = (const float*)d_in[17];

    char* ws = (char*)d_ws;
    size_t off = 0;
    auto alloc = [&](size_t bytes) -> void* {
        void* p = ws + off;
        off = (off + bytes + 255) & ~(size_t)255;
        return p;
    };

    __bf16* x_bf    = (__bf16*)alloc((size_t)Bc * Tc * Dc * 2);   // 16 MB
    __bf16* wihf_bf = (__bf16*)alloc((size_t)FHc * Dc * 2);
    __bf16* whhf_bf = (__bf16*)alloc((size_t)FHc * Hc * 2);
    __bf16* wihb_bf = (__bf16*)alloc((size_t)FHc * Dc * 2);
    __bf16* whhb_bf = (__bf16*)alloc((size_t)FHc * Hc * 2);
    float*  Xp_f    = (float*)alloc((size_t)Tc * Bc * FHc * 4);   // 134 MB
    float*  Xp_b    = (float*)alloc((size_t)Tc * Bc * FHc * 4);   // 134 MB
    float*  hf      = (float*)alloc((size_t)Bc * Tc * Hc * 4);    // 33.5 MB
    float*  hb      = (float*)alloc((size_t)Bc * Tc * Hc * 4);    // 33.5 MB
    float*  emis    = (float*)alloc((size_t)Bc * Tc * Kc * 4);
    float*  partial = (float*)alloc((size_t)Bc * 4);
    unsigned char* hist = (unsigned char*)alloc((size_t)Bc * Tc * Kc);

    float* out = (float*)d_out;

    // 1) cast weights fp32 -> bf16
    cast_bf16_k<<<(FHc * Dc + 255) / 256, 256, 0, stream>>>(Wih_f, wihf_bf, FHc * Dc);
    cast_bf16_k<<<(FHc * Hc + 255) / 256, 256, 0, stream>>>(Whh_f, whhf_bf, FHc * Hc);
    cast_bf16_k<<<(FHc * Dc + 255) / 256, 256, 0, stream>>>(Wih_b, wihb_bf, FHc * Dc);
    cast_bf16_k<<<(FHc * Hc + 255) / 256, 256, 0, stream>>>(Whh_b, whhb_bf, FHc * Hc);

    // 2) embedding gather -> bf16
    embed_k<<<(Bc * Tc * Dc) / 256, 256, 0, stream>>>(input_x, emb, x_bf);

    // 3) input-projection GEMMs (both directions), WMMA bf16
    inproj_gemm_k<<<dim3((Bc * Tc) / 16, 2), 256, 0, stream>>>(
        x_bf, wihf_bf, b_f, wihb_bf, b_b, Xp_f, Xp_b);

    // 4) recurrent LSTM (both directions); Whh register/LDS-resident
    constexpr size_t lstm_lds = (size_t)16 * Hc * 2 + (size_t)FHc * 128 * 2; // 264KB
    lstm_k<<<dim3(Bc / 16, 2), 256, lstm_lds, stream>>>(
        Xp_f, whhf_bf, h0f, c0f, hf,
        Xp_b, whhb_bf, h0b, c0b, hb);

    // 5) emissions
    emis_k<<<(Bc * Tc * Kc + 255) / 256, 256, 0, stream>>>(hf, hb, W_emit, b_emit, emis);

    // 6) CRF loss + deterministic reduction
    crf_loss_k<<<1, 256, 0, stream>>>(emis, label, input_x, start_t, end_t, trans, partial);
    reduce_k<<<1, 256, 0, stream>>>(partial, out);

    // 7) Viterbi decode -> out[1..]
    viterbi_k<<<1, 256, 0, stream>>>(emis, start_t, end_t, trans, hist, out + 1);
}
